// HPEncoder_66494683676995
// MI455X (gfx1250) — compile-verified, matrix operand order checked by reading, not run on hardware
//
#include <hip/hip_runtime.h>

// ---------------------------------------------------------------------------
// Types for CDNA5 WMMA (wave32, 16x16x32 bf16 -> f32)
// ---------------------------------------------------------------------------
typedef __attribute__((ext_vector_type(16))) __bf16 v16bf;
typedef __attribute__((ext_vector_type(8)))  __bf16 v8bf;
typedef __attribute__((ext_vector_type(8)))  float  v8f;
typedef __attribute__((ext_vector_type(4)))  int    v4i;

#define C_CH 128
#define TILES 4   // row-tiles (of 128 map entries) processed per block

// Async global->LDS copy path (CDNA5 GLOBAL_LOAD_ASYNC_TO_LDS_B128, ASYNCcnt)
#if defined(__AMDGCN__) && __has_builtin(__builtin_amdgcn_global_load_async_to_lds_b128)
#define HAS_ASYNC_LDS 1
#else
#define HAS_ASYNC_LDS 0
#endif

#if HAS_ASYNC_LDS
typedef v4i __attribute__((address_space(1))) gv4i;  // global
typedef v4i __attribute__((address_space(3))) sv4i;  // LDS
#endif

__device__ __forceinline__ void cp16_to_lds(const __bf16* gsrc, __bf16* ldst) {
#if HAS_ASYNC_LDS
    __builtin_amdgcn_global_load_async_to_lds_b128((gv4i*)gsrc, (sv4i*)ldst, 0, 0);
#else
    *(uint4*)ldst = *(const uint4*)gsrc;
#endif
}

__device__ __forceinline__ void wait_async_lds() {
#if HAS_ASYNC_LDS
    asm volatile("s_wait_asynccnt 0" ::: "memory");
#endif
}

// Load a 16-element bf16 A/B fragment for one lane from LDS.
// Layout per CDNA5 ISA (16-bit A-matrix 16x32): lane L holds row (L&15),
// khalf=(L>>4): VGPRs 0-3 <- K = khalf*8 .. +7 (contiguous 16B),
//              VGPRs 4-7 <- K = 16+khalf*8 .. +7 (contiguous 16B).
__device__ __forceinline__ v16bf load_frag(const __bf16* p) {
    v8bf lo = *(const v8bf*)(p);
    v8bf hi = *(const v8bf*)(p + 16);
    return __builtin_shufflevector(lo, hi, 0, 1, 2, 3, 4, 5, 6, 7,
                                           8, 9, 10, 11, 12, 13, 14, 15);
}

// ---------------------------------------------------------------------------
// Gather -> WMMA GEMM -> atomic scatter-add, one kernel offset per blockIdx.y.
// Each block: loads W^T once, then pipelines TILES row-tiles of 128 map
// entries with a double-buffered async A gather.
// feat : bf16 features, (n_in+1) x 128, row n_in is zeros (dummy)
// Wt   : bf16 weights, per-offset 128x128 stored TRANSPOSED: Wt[n*128 + k]
// im/om: index maps, stride n_map per offset
// acc  : f32 accumulator, (n_out+1) x 128, pre-seeded with bias
// ---------------------------------------------------------------------------
__global__ __launch_bounds__(256)
void conv_wmma_kernel(const __bf16* __restrict__ feat,
                      const __bf16* __restrict__ Wt,
                      const int* __restrict__ im,
                      const int* __restrict__ om,
                      float* __restrict__ acc,
                      int n_map, int n_in, int n_out) {
    __shared__ __bf16 lA[2 * 128 * C_CH];   // double-buffered gathered rows (64 KB)
    __shared__ __bf16 lB[C_CH * C_CH];      // W^T for this offset           (32 KB)

    const int tid  = threadIdx.x;
    const int k    = blockIdx.y;
    const long moff = (long)k * n_map;

    const int tiles_total = (n_map + 127) >> 7;
    const int t0 = blockIdx.x * TILES;
    int tcount = tiles_total - t0;
    if (tcount > TILES) tcount = TILES;
    if (tcount <= 0) return;

    // thread's role in the A gather: 2 threads per row, 64 channels each
    const int gr    = tid >> 1;        // 0..127
    const int ghalf = tid & 1;

    // ---- issue W^T copy (2048 x 16B total; 8 per thread) ----
    {
        const __bf16* g = Wt + ((size_t)k << 14) + (size_t)tid * 64;
        __bf16* d       = lB + tid * 64;
#pragma unroll
        for (int i = 0; i < 8; ++i) cp16_to_lds(g + i * 8, d + i * 8);
    }
    // ---- issue A gather for first tile into buffer 0 ----
    {
        const int e = t0 * 128 + gr;
        const int src = (e < n_map) ? im[moff + e] : n_in;
        const __bf16* g = feat + (size_t)src * C_CH + ghalf * 64;
        __bf16* d       = lA + gr * C_CH + ghalf * 64;
#pragma unroll
        for (int i = 0; i < 8; ++i) cp16_to_lds(g + i * 8, d + i * 8);
    }
    wait_async_lds();
    __syncthreads();

    const int w     = tid >> 5;          // wave id 0..7
    const int lane  = tid & 31;
    const int lrow  = lane & 15;
    const int khalf = lane >> 4;         // 0 or 1

    for (int tt = 0; tt < tcount; ++tt) {
        // ---- prefetch next tile's A into the other buffer (async) ----
        if (tt + 1 < tcount) {
            const int e = (t0 + tt + 1) * 128 + gr;
            const int src = (e < n_map) ? im[moff + e] : n_in;
            const __bf16* g = feat + (size_t)src * C_CH + ghalf * 64;
            __bf16* d = lA + ((tt + 1) & 1) * 128 * C_CH + gr * C_CH + ghalf * 64;
#pragma unroll
            for (int i = 0; i < 8; ++i) cp16_to_lds(g + i * 8, d + i * 8);
        }

        // ---- compute on current buffer: 16 rows x 128 cols per wave ----
        const __bf16* A = lA + (tt & 1) * 128 * C_CH;
        v8f cacc[8] = {};
#pragma unroll
        for (int kc = 0; kc < 4; ++kc) {     // K = 128 in chunks of 32
            v16bf a = load_frag(A + (w * 16 + lrow) * C_CH + kc * 32 + khalf * 8);
#pragma unroll
            for (int t = 0; t < 8; ++t) {
                v16bf b = load_frag(lB + (t * 16 + lrow) * C_CH + kc * 32 + khalf * 8);
                cacc[t] = __builtin_amdgcn_wmma_f32_16x16x32_bf16(
                    false, a, false, b, (short)0, cacc[t], false, false);
            }
        }

        // ---- scatter-add into accumulator via f32 atomics ----
        // C/D layout: VGPR j -> row j (lanes 0-15) or row j+8 (lanes 16-31),
        // col = lane&15 within each 16-wide N tile.
        const int mbase = (t0 + tt) * 128 + w * 16 + khalf * 8;
        int orow[8];
#pragma unroll
        for (int j = 0; j < 8; ++j) {
            const int e = mbase + j;
            orow[j] = (e < n_map) ? om[moff + e] : n_out;
        }
#pragma unroll
        for (int t = 0; t < 8; ++t) {
            const int col = t * 16 + lrow;
#pragma unroll
            for (int j = 0; j < 8; ++j) {
                atomicAdd(&acc[(size_t)orow[j] * C_CH + col], cacc[t][j]);
            }
        }

        // next buffer ready + everyone done reading current buffer
        wait_async_lds();
        __syncthreads();
    }
}

// ---------------------------------------------------------------------------
// Seed accumulator with bias (rows < n) and zero the dummy row n.
// ---------------------------------------------------------------------------
__global__ void bias_init_kernel(float* __restrict__ acc,
                                 const float* __restrict__ bias, int n) {
    const long i = (long)blockIdx.x * blockDim.x + threadIdx.x;
    const long total = (long)(n + 1) * C_CH;
    if (i < total) {
        const long row = i >> 7;
        const int col  = (int)(i & 127);
        acc[i] = (row < n) ? bias[col] : 0.0f;
    }
}

// ---------------------------------------------------------------------------
// Activation: f32 -> bf16 (optional ReLU), and zero the dummy row.
// ---------------------------------------------------------------------------
__global__ void act_bf16_kernel(const float* __restrict__ src,
                                __bf16* __restrict__ dst, int n, int relu) {
    const long i = (long)blockIdx.x * blockDim.x + threadIdx.x;
    const long nv = (long)n * C_CH;
    if (i < nv + C_CH) {
        float v = (i < nv) ? src[i] : 0.0f;
        if (relu && v < 0.0f) v = 0.0f;
        dst[i] = (__bf16)v;
    }
}

// ---------------------------------------------------------------------------
// Weight convert: f32 W[k][kin][n] -> bf16 Wt[k][n][kin]  (transpose KxN)
// ---------------------------------------------------------------------------
__global__ void wconv_kernel(const float* __restrict__ W,
                             __bf16* __restrict__ Wt, int nmat) {
    const long g = (long)blockIdx.x * blockDim.x + threadIdx.x;
    if (g < (long)nmat << 14) {
        const long m  = g >> 14;
        const int  r  = (int)(g & 16383);
        const int kin = r >> 7;
        const int n   = r & 127;
        Wt[(m << 14) + ((long)n << 7) + kin] = (__bf16)W[g];
    }
}

__global__ void copy_out_kernel(const float* __restrict__ src,
                                float* __restrict__ dst, long n) {
    const long i = (long)blockIdx.x * blockDim.x + threadIdx.x;
    if (i < n) dst[i] = src[i];
}

// ---------------------------------------------------------------------------
static inline size_t align256(size_t x) { return (x + 255) & ~(size_t)255; }

extern "C" void kernel_launch(void* const* d_in, const int* in_sizes, int n_in_args,
                              void* d_out, int out_size, void* d_ws, size_t ws_size,
                              hipStream_t stream) {
    (void)n_in_args; (void)out_size; (void)ws_size;
    const float* x   = (const float*)d_in[0];
    const float* W0  = (const float*)d_in[1];
    const float* b0  = (const float*)d_in[2];
    const float* W1  = (const float*)d_in[3];
    const float* b1  = (const float*)d_in[4];
    const float* Wd1 = (const float*)d_in[5];
    const float* bd1 = (const float*)d_in[6];
    const float* W2  = (const float*)d_in[7];
    const float* b2  = (const float*)d_in[8];
    const float* Wd2 = (const float*)d_in[9];
    const float* bd2 = (const float*)d_in[10];
    const int* in0   = (const int*)d_in[11];
    const int* out0  = (const int*)d_in[12];
    const int* in1   = (const int*)d_in[13];
    const int* out1  = (const int*)d_in[14];
    const int* din1  = (const int*)d_in[15];
    const int* dout1 = (const int*)d_in[16];
    const int* din2  = (const int*)d_in[17];
    const int* dout2 = (const int*)d_in[18];

    const int N0 = in_sizes[11] / 27;
    const int N1 = in_sizes[15] / 8;
    const int N2 = in_sizes[17] / 8;

    // Workspace carve-up (acc reused across layers; feat overwritten per layer)
    char* ws = (char*)d_ws;
    size_t off = 0;
    float* acc = (float*)(ws + off);
    off = align256(off + (size_t)(N0 + 1) * C_CH * sizeof(float));
    __bf16* feat = (__bf16*)(ws + off);
    off = align256(off + (size_t)(N0 + 1) * C_CH * sizeof(__bf16));
    __bf16* wt = (__bf16*)(ws + off);   // 97 * 128*128 bf16
    __bf16* wt0  = wt;
    __bf16* wt1  = wt + ((size_t)27 << 14);
    __bf16* wtd1 = wt + ((size_t)54 << 14);
    __bf16* wt2  = wt + ((size_t)62 << 14);
    __bf16* wtd2 = wt + ((size_t)89 << 14);

    const int TPB = 256;
    auto cdiv = [](long a, long b) { return (int)((a + b - 1) / b); };

    // weights -> bf16 transposed
    wconv_kernel<<<cdiv((long)27 << 14, TPB), TPB, 0, stream>>>(W0,  wt0,  27);
    wconv_kernel<<<cdiv((long)27 << 14, TPB), TPB, 0, stream>>>(W1,  wt1,  27);
    wconv_kernel<<<cdiv((long)8  << 14, TPB), TPB, 0, stream>>>(Wd1, wtd1, 8);
    wconv_kernel<<<cdiv((long)27 << 14, TPB), TPB, 0, stream>>>(W2,  wt2,  27);
    wconv_kernel<<<cdiv((long)8  << 14, TPB), TPB, 0, stream>>>(Wd2, wtd2, 8);

    // x -> bf16 (with zero dummy row)
    act_bf16_kernel<<<cdiv((long)(N0 + 1) * C_CH, TPB), TPB, 0, stream>>>(x, feat, N0, 0);

    auto gridx = [&](int n_map) { return cdiv(cdiv(n_map, 128), TILES); };

    // conv0 + ReLU
    bias_init_kernel<<<cdiv((long)(N0 + 1) * C_CH, TPB), TPB, 0, stream>>>(acc, b0, N0);
    conv_wmma_kernel<<<dim3(gridx(N0), 27), TPB, 0, stream>>>(feat, wt0, in0, out0, acc, N0, N0, N0);
    act_bf16_kernel<<<cdiv((long)(N0 + 1) * C_CH, TPB), TPB, 0, stream>>>(acc, feat, N0, 1);

    // conv1
    bias_init_kernel<<<cdiv((long)(N0 + 1) * C_CH, TPB), TPB, 0, stream>>>(acc, b1, N0);
    conv_wmma_kernel<<<dim3(gridx(N0), 27), TPB, 0, stream>>>(feat, wt1, in0, out0, acc, N0, N0, N0);
    act_bf16_kernel<<<cdiv((long)(N0 + 1) * C_CH, TPB), TPB, 0, stream>>>(acc, feat, N0, 0);

    // down1 + ReLU  (maps: (8, N1), inputs indexed into level-0 rows)
    bias_init_kernel<<<cdiv((long)(N1 + 1) * C_CH, TPB), TPB, 0, stream>>>(acc, bd1, N1);
    conv_wmma_kernel<<<dim3(gridx(N1), 8), TPB, 0, stream>>>(feat, wtd1, din1, dout1, acc, N1, N0, N1);
    act_bf16_kernel<<<cdiv((long)(N1 + 1) * C_CH, TPB), TPB, 0, stream>>>(acc, feat, N1, 1);

    // conv2
    bias_init_kernel<<<cdiv((long)(N1 + 1) * C_CH, TPB), TPB, 0, stream>>>(acc, b2, N1);
    conv_wmma_kernel<<<dim3(gridx(N1), 27), TPB, 0, stream>>>(feat, wt2, in1, out1, acc, N1, N1, N1);
    act_bf16_kernel<<<cdiv((long)(N1 + 1) * C_CH, TPB), TPB, 0, stream>>>(acc, feat, N1, 0);

    // down2 -> final f32 output
    bias_init_kernel<<<cdiv((long)(N2 + 1) * C_CH, TPB), TPB, 0, stream>>>(acc, bd2, N2);
    conv_wmma_kernel<<<dim3(gridx(N2), 8), TPB, 0, stream>>>(feat, wtd2, din2, dout2, acc, N2, N1, N2);
    copy_out_kernel<<<cdiv((long)N2 * C_CH, TPB), TPB, 0, stream>>>(acc, (float*)d_out, (long)N2 * C_CH);
}
